// ISA_Block_88493506167125
// MI455X (gfx1250) — compile-verified, matrix operand order checked by reading, not run on hardware
//
#include <hip/hip_runtime.h>
#include <hip/hip_bf16.h>
#include <math.h>

// ---------------------------------------------------------------------------
// MI455X (gfx1250) implementation of the two-stage windowed MHA block.
// All GEMMs and both attention stages run through v_wmma_f32_16x16x32_f16
// (wave32, 16x16 f32 accum). Intermediates are f16 in d_ws to halve HBM
// traffic (~25 GFLOP vs ~100MB moved -> memory/compute balanced on MI455X).
// GEMM tile staging uses GLOBAL_LOAD_ASYNC_TO_LDS_B128 (ASYNCcnt) when the
// toolchain exposes the builtin, plus global_prefetch for the next K chunk.
// ---------------------------------------------------------------------------

typedef _Float16 h16;
typedef __attribute__((ext_vector_type(16))) _Float16 v16h;
typedef __attribute__((ext_vector_type(8)))  float    v8f;

#define TOK 16384      // tokens in both layouts (t*bnh*bnw * n*HB*WB)
#define CDIM 256
#define ST 40          // LDS h16 row stride: 80B, 16B aligned

// ---- gfx1250 async Global->LDS path (guarded; fallback always compiles) ---
#if defined(__has_builtin)
#  if __has_builtin(__builtin_amdgcn_global_load_async_to_lds_b128)
#    define USE_ASYNC_LDS 1
#  endif
#endif
#ifndef USE_ASYNC_LDS
#  define USE_ASYNC_LDS 0
#endif

// exact parameter types per hipcc diagnostic: vector_size(16) int, AS1 / AS3
typedef int v4i_t __attribute__((vector_size(16)));
typedef __attribute__((address_space(1))) v4i_t as1_v4i;
typedef __attribute__((address_space(3))) v4i_t as3_v4i;

__device__ __forceinline__ as1_v4i* as_gv4(const void* p) {
  return (as1_v4i*)(unsigned long long)(uintptr_t)p;
}
__device__ __forceinline__ as3_v4i* as_lv4(void* p) {
  // generic LDS pointers carry the LDS byte offset in the low 32 bits
  return (as3_v4i*)(unsigned int)(uintptr_t)p;
}

__device__ __forceinline__ void wait_async0() {
#if defined(__has_builtin)
#  if __has_builtin(__builtin_amdgcn_s_wait_asynccnt)
  __builtin_amdgcn_s_wait_asynccnt(0);
#  else
  asm volatile("s_wait_asynccnt 0x0" ::: "memory");
#  endif
#else
  asm volatile("s_wait_asynccnt 0x0" ::: "memory");
#endif
}

// ---- WMMA helpers ---------------------------------------------------------

__device__ __forceinline__ v8f wmma_f16(v16h a, v16h b, v8f c) {
  // D = A(16x32 f16) x B(32x16 f16) + C(16x16 f32)
  return __builtin_amdgcn_wmma_f32_16x16x32_f16(
      /*neg_a=*/false, a, /*neg_b=*/false, b,
      /*c_mod=*/(short)0, c, /*reuse_a=*/false, /*reuse_b=*/false);
}

// Load one 16x32 f16 tile from LDS in A-fragment layout (ISA 7.12.2):
// lanes 0-15 hold row=lane, K 0-7 & 16-23; lanes 16-31 hold row=lane-16,
// K 8-15 & 24-31. Loading B^T rows with this helper yields the B fragment.
__device__ __forceinline__ v16h load_frag(const h16* p, int st) {
  int lane = threadIdx.x & 31;
  int row  = lane & 15;
  int ko   = (lane >> 4) << 3;     // 0 or 8
  const h16* q = p + row * st + ko;
  v16h f;
#pragma unroll
  for (int i = 0; i < 8; ++i) f[i] = q[i];
#pragma unroll
  for (int i = 0; i < 8; ++i) f[8 + i] = q[16 + i];
  return f;
}

// Store 16x16 f32 accumulator tile to LDS/f32 buffer (C/D layout):
// lane<16: N=lane, M=vgpr; lane>=16: N=lane-16, M=8+vgpr.
__device__ __forceinline__ void store_tile(float* dst, int ld, v8f acc, float s) {
  int lane = threadIdx.x & 31;
  int n = lane & 15, mb = (lane >> 4) << 3;
#pragma unroll
  for (int r = 0; r < 8; ++r) dst[(mb + r) * ld + n] = acc[r] * s;
}

// ---- elementwise / layout kernels ----------------------------------------

__global__ void k_f32_to_f16(const float* __restrict__ src, h16* __restrict__ dst, int n) {
  int i = blockIdx.x * 256 + threadIdx.x;
  if (i < n) dst[i] = (h16)src[i];
}

// to_long gather: [n,c,t,h,w] -> long rows r1 = l1*512+b1 ; build q+pos, k+pos, v, pos
__global__ void k_pack_long(const float* __restrict__ q, const float* __restrict__ k,
                            const float* __restrict__ v, const float* __restrict__ pos,
                            h16* __restrict__ Xq, h16* __restrict__ Xk,
                            h16* __restrict__ Xv, h16* __restrict__ Pl) {
  int idx = blockIdx.x * 256 + threadIdx.x;          // over TOK*CDIM
  int r1 = idx >> 8, c = idx & 255;
  int l1 = r1 >> 9, b1 = r1 & 511;
  int tt = l1 >> 4, by = (l1 >> 2) & 3, bx = l1 & 3;
  int nn = b1 >> 8, iy = (b1 >> 4) & 15, ix = b1 & 15;
  int y = by * 16 + iy, x = bx * 16 + ix;
  size_t src = ((((size_t)nn * 256 + c) * 2 + tt) * 64 + y) * 64 + x;
  float pv = pos[src];
  Xq[idx] = (h16)(q[src] + pv);
  Xk[idx] = (h16)(k[src] + pv);
  Xv[idx] = (h16)v[src];
  Pl[idx] = (h16)pv;
}

// long_to_short remap: build Xq2 = v1 + pos (short order), Xv2 = v1 (short order)
__global__ void k_long_to_short(const h16* __restrict__ v1, const h16* __restrict__ Pl,
                                h16* __restrict__ Xq2, h16* __restrict__ Xv2) {
  int idx = blockIdx.x * 256 + threadIdx.x;
  int r1 = idx >> 8, c = idx & 255;
  int l1 = r1 >> 9, b1 = r1 & 511;
  int tt = l1 >> 4, by = (l1 >> 2) & 3, bx = l1 & 3;
  int nn = b1 >> 8, iy = (b1 >> 4) & 15, ix = b1 & 15;
  int r2 = (((tt * 16 + iy) * 16 + ix) << 5) + ((nn * 4 + by) * 4 + bx);
  float a = (float)v1[idx], p = (float)Pl[idx];
  Xq2[(size_t)r2 * 256 + c] = (h16)(a + p);
  Xv2[(size_t)r2 * 256 + c] = v1[idx];
}

// short rows -> [n,c,t,h,w] f32 output
__global__ void k_scatter(const float* __restrict__ Y, float* __restrict__ out) {
  int idx = blockIdx.x * 256 + threadIdx.x;
  int r2 = idx >> 8, c = idx & 255;
  int l2 = r2 >> 5, b2 = r2 & 31;
  int tt = l2 >> 8, iy = (l2 >> 4) & 15, ix = l2 & 15;
  int nn = b2 >> 4, by = (b2 >> 2) & 3, bx = b2 & 3;
  int y = by * 16 + iy, x = bx * 16 + ix;
  out[((((size_t)nn * 256 + c) * 2 + tt) * 64 + y) * 64 + x] = Y[idx];
}

// ---- WMMA GEMM: C[M,N] = A[M,K] * W[N,K]^T + bias, A/W f16 row-major ------
// 256 threads = 8 waves; block tile 64x64; wave tile 16(M)x32(N); K step 32.
// Tiles staged to LDS via async Global->LDS b128 when available.
template <typename OutT>
__global__ void __launch_bounds__(256) k_gemm(const h16* __restrict__ A,
                                              const h16* __restrict__ W,
                                              const float* __restrict__ bias,
                                              OutT* __restrict__ C,
                                              int M, int N, int K) {
  __shared__ h16 As[64 * ST];
  __shared__ h16 Bs[64 * ST];
  int tid = threadIdx.x;
  int wid = tid >> 5, lane = tid & 31;
  int wm = wid & 3, wn = wid >> 2;
  int m0 = blockIdx.y * 64, n0 = blockIdx.x * 64;
  int lr = tid >> 2;            // 0..63
  int lc = (tid & 3) << 3;      // 0,8,16,24
  h16* la = &As[lr * ST + lc];
  h16* lb = &Bs[lr * ST + lc];
  v8f acc0 = {}, acc1 = {};
  for (int k0 = 0; k0 < K; k0 += 32) {
    const h16* ga = A + (size_t)(m0 + lr) * K + k0 + lc;
    const h16* gb = W + (size_t)(n0 + lr) * K + k0 + lc;
    if (k0 + 32 < K) {                         // prefetch next K chunk
      __builtin_prefetch(ga + 32, 0, 0);
      __builtin_prefetch(gb + 32, 0, 0);
    }
#if USE_ASYNC_LDS
    __builtin_amdgcn_global_load_async_to_lds_b128(as_gv4(ga), as_lv4(la), 0, 0);
    __builtin_amdgcn_global_load_async_to_lds_b128(as_gv4(gb), as_lv4(lb), 0, 0);
    wait_async0();
#else
#pragma unroll
    for (int i = 0; i < 8; ++i) la[i] = ga[i];
#pragma unroll
    for (int i = 0; i < 8; ++i) lb[i] = gb[i];
#endif
    __syncthreads();
    v16h a  = load_frag(As + (wm * 16) * ST, ST);
    v16h b0 = load_frag(Bs + (wn * 32) * ST, ST);
    v16h b1 = load_frag(Bs + (wn * 32 + 16) * ST, ST);
    acc0 = wmma_f16(a, b0, acc0);
    acc1 = wmma_f16(a, b1, acc1);
    __syncthreads();
  }
  int n = lane & 15, mb = (lane >> 4) << 3;
  int gm = m0 + wm * 16 + mb;
  int gn = n0 + wn * 32;
  float bv0 = bias[gn + n], bv1 = bias[gn + 16 + n];
#pragma unroll
  for (int r = 0; r < 8; ++r) {
    C[(size_t)(gm + r) * N + gn + n]      = (OutT)(acc0[r] + bv0);
    C[(size_t)(gm + r) * N + gn + 16 + n] = (OutT)(acc1[r] + bv1);
  }
}

// ---- Attention stage 1: 4096 problems, L=32, d=32, one wave each ----------
__global__ void __launch_bounds__(64) k_attn1(const h16* __restrict__ Q1,
                                              const h16* __restrict__ K1,
                                              const h16* __restrict__ V1,
                                              h16* __restrict__ O1) {
  __shared__ h16  sh[2][4 * 32 * ST];   // per wave: Qs, Ks, Vt, Ps
  __shared__ float ss[2][32 * 32];      // per wave: scores f32
  int wid = threadIdx.x >> 5, lane = threadIdx.x & 31;
  int prob = blockIdx.x * 2 + wid;
  int bb = prob >> 3, hh = prob & 7;
  int fh = hh * 32;
  h16* Qs = sh[wid];
  h16* Ks = Qs + 32 * ST;
  h16* Vt = Ks + 32 * ST;
  h16* Ps = Vt + 32 * ST;
  float* S = ss[wid];

  int l = lane;                                   // token row this lane loads
  size_t base = ((size_t)(l * 512 + bb)) * 256 + fh;
#pragma unroll 4
  for (int c = 0; c < 32; ++c) {
    Qs[l * ST + c] = Q1[base + c];
    Ks[l * ST + c] = K1[base + c];
    Vt[c * ST + l] = V1[base + c];               // V^T for B fragments
  }
  __syncthreads();

  v16h a0 = load_frag(Qs, ST),       a1 = load_frag(Qs + 16 * ST, ST);
  v16h k0 = load_frag(Ks, ST),       k1 = load_frag(Ks + 16 * ST, ST);
  v8f z = {};
  v8f s00 = wmma_f16(a0, k0, z), s01 = wmma_f16(a0, k1, z);
  v8f s10 = wmma_f16(a1, k0, z), s11 = wmma_f16(a1, k1, z);
  const float scale = 0.17677669529663687f;       // 1/sqrt(32)
  store_tile(S + 0 * 32 + 0,  32, s00, scale);
  store_tile(S + 0 * 32 + 16, 32, s01, scale);
  store_tile(S + 16 * 32 + 0, 32, s10, scale);
  store_tile(S + 16 * 32 + 16,32, s11, scale);
  __syncthreads();

  // softmax: each lane owns one row of 32
  float mx = -1e30f;
#pragma unroll 4
  for (int j = 0; j < 32; ++j) mx = fmaxf(mx, S[l * 32 + j]);
  float sum = 0.f;
#pragma unroll 4
  for (int j = 0; j < 32; ++j) { float e = __expf(S[l * 32 + j] - mx); S[l * 32 + j] = e; sum += e; }
  float inv = 1.f / sum;
#pragma unroll 4
  for (int j = 0; j < 32; ++j) Ps[l * ST + j] = (h16)(S[l * 32 + j] * inv);
  __syncthreads();

  v16h p0 = load_frag(Ps, ST),      p1 = load_frag(Ps + 16 * ST, ST);
  v16h v0 = load_frag(Vt, ST),      v1 = load_frag(Vt + 16 * ST, ST);
  v8f o00 = wmma_f16(p0, v0, z), o01 = wmma_f16(p0, v1, z);
  v8f o10 = wmma_f16(p1, v0, z), o11 = wmma_f16(p1, v1, z);

  int n = lane & 15, mb = (lane >> 4) << 3;
#pragma unroll
  for (int r = 0; r < 8; ++r) {
    int lq0 = mb + r, lq1 = 16 + mb + r;
    size_t o0 = ((size_t)(lq0 * 512 + bb)) * 256 + fh;
    size_t o1 = ((size_t)(lq1 * 512 + bb)) * 256 + fh;
    O1[o0 + n]      = (h16)o00[r];
    O1[o0 + 16 + n] = (h16)o01[r];
    O1[o1 + n]      = (h16)o10[r];
    O1[o1 + 16 + n] = (h16)o11[r];
  }
}

// ---- Attention stage 2: 256 problems, L=512, d=32. One WG (4 waves) per
// (batch,head,query-block-of-16). Scores 16x512 in LDS, 2-pass softmax,
// per-wave K-chunk accumulation for P*V, LDS reduce. ------------------------
__global__ void __launch_bounds__(128) k_attn2(const h16* __restrict__ Q2,
                                               const h16* __restrict__ K2,
                                               const h16* __restrict__ V2,
                                               h16* __restrict__ O2) {
  __shared__ float S[16 * 512];          // scores (later reused as partials)
  __shared__ h16   P[16 * 512];          // unnormalized probabilities
  __shared__ h16   Qs[16 * ST];
  __shared__ h16   WT[4][32 * ST];       // per-wave K/V staging
  __shared__ float rowmax[16], rowsum[16], red[128];

  int tid = threadIdx.x, wid = tid >> 5, lane = tid & 31;
  int idx = blockIdx.x;
  int qb = idx & 31, hh = (idx >> 5) & 7, b2 = idx >> 8;
  int fh = hh * 32;

  for (int e = tid; e < 512; e += 128) {          // load 16 query rows
    int m = e >> 5, c = e & 31;
    int l = qb * 16 + m;
    Qs[m * ST + c] = Q2[((size_t)(l * 32 + b2)) * 256 + fh + c];
  }
  __syncthreads();

  const float scale = 0.17677669529663687f;
  v8f z = {};
  for (int it = 0; it < 8; ++it) {                // 32 key blocks / 4 waves
    int kb = wid + it * 4;
    int row = lane & 15, co = (lane >> 4) * 16;
    const h16* kp = K2 + ((size_t)((kb * 16 + row) * 32 + b2)) * 256 + fh + co;
#pragma unroll
    for (int i = 0; i < 16; ++i) WT[wid][row * ST + co + i] = kp[i];
    __syncthreads();
    v16h a = load_frag(Qs, ST);
    v16h b = load_frag(WT[wid], ST);
    v8f s = wmma_f16(a, b, z);
    int n = lane & 15, mb = (lane >> 4) << 3;
#pragma unroll
    for (int r = 0; r < 8; ++r) S[(mb + r) * 512 + kb * 16 + n] = s[r] * scale;
    __syncthreads();
  }

  // two-pass softmax over 512 per row; 8 threads x 64 cols per row
  int row = tid & 15, ch = tid >> 4;
  float mx = -1e30f;
  for (int j = ch * 64; j < ch * 64 + 64; ++j) mx = fmaxf(mx, S[row * 512 + j]);
  red[tid] = mx;
  __syncthreads();
  if (tid < 16) {
    float m2 = red[tid];
#pragma unroll
    for (int c = 1; c < 8; ++c) m2 = fmaxf(m2, red[c * 16 + tid]);
    rowmax[tid] = m2;
  }
  __syncthreads();
  float rm = rowmax[row];
  float sum = 0.f;
  for (int j = ch * 64; j < ch * 64 + 64; ++j) {
    float e = __expf(S[row * 512 + j] - rm);
    P[row * 512 + j] = (h16)e;                    // unnormalized; divide later
    sum += e;
  }
  red[tid] = sum;
  __syncthreads();
  if (tid < 16) {
    float s2 = 0.f;
#pragma unroll
    for (int c = 0; c < 8; ++c) s2 += red[c * 16 + tid];
    rowsum[tid] = s2;
  }
  __syncthreads();

  // P @ V: each wave accumulates 4 chunks of 32 keys
  v8f acc0 = {}, acc1 = {};
  for (int it = 0; it < 4; ++it) {
    int kc = wid + it * 4;
    const h16* vp = V2 + ((size_t)((kc * 32 + lane) * 32 + b2)) * 256 + fh;
#pragma unroll 4
    for (int c = 0; c < 32; ++c) WT[wid][c * ST + lane] = vp[c];   // V^T
    __syncthreads();
    v16h a  = load_frag(P + kc * 32, 512);
    v16h b0 = load_frag(WT[wid], ST);
    v16h b1 = load_frag(WT[wid] + 16 * ST, ST);
    acc0 = wmma_f16(a, b0, acc0);
    acc1 = wmma_f16(a, b1, acc1);
    __syncthreads();
  }
  float* part = S;                                 // reuse score LDS
  {
    int n = lane & 15, mb = (lane >> 4) << 3;
#pragma unroll
    for (int r = 0; r < 8; ++r) {
      part[wid * 512 + (mb + r) * 32 + n]      = acc0[r];
      part[wid * 512 + (mb + r) * 32 + 16 + n] = acc1[r];
    }
  }
  __syncthreads();
  for (int e = tid; e < 512; e += 128) {
    int m = e >> 5, n = e & 31;
    float v = part[m * 32 + n] + part[512 + m * 32 + n] +
              part[1024 + m * 32 + n] + part[1536 + m * 32 + n];
    v /= rowsum[m];
    int l = qb * 16 + m;
    O2[((size_t)(l * 32 + b2)) * 256 + fh + n] = (h16)v;
  }
}

// ---------------------------------------------------------------------------

extern "C" void kernel_launch(void* const* d_in, const int* in_sizes, int n_in,
                              void* d_out, int out_size, void* d_ws, size_t ws_size,
                              hipStream_t stream) {
  (void)in_sizes; (void)n_in; (void)out_size; (void)ws_size;
  const float* q      = (const float*)d_in[0];
  const float* k      = (const float*)d_in[1];
  const float* v      = (const float*)d_in[2];
  const float* pos    = (const float*)d_in[3];
  const float* wl_in  = (const float*)d_in[4];
  const float* bl_in  = (const float*)d_in[5];
  const float* wl_out = (const float*)d_in[6];
  const float* bl_out = (const float*)d_in[7];
  const float* ws_in  = (const float*)d_in[8];
  const float* bs_in  = (const float*)d_in[9];
  const float* ws_out = (const float*)d_in[10];
  const float* bs_out = (const float*)d_in[11];
  float* out = (float*)d_out;

  const size_t BUFB = (size_t)TOK * CDIM * sizeof(h16);   // 8 MB each
  char* ws = (char*)d_ws;
  h16* B[9];
  for (int i = 0; i < 9; ++i) B[i] = (h16*)(ws + i * BUFB);
  float* Y2       = (float*)(ws + 9 * BUFB);
  h16*  wl_in_h   = (h16*)((char*)Y2 + (size_t)TOK * CDIM * sizeof(float));
  h16*  wl_out_h  = wl_in_h + 768 * 256;
  h16*  ws_in_h   = wl_out_h + 256 * 256;
  h16*  ws_out_h  = ws_in_h + 768 * 256;

  // weight conversion
  k_f32_to_f16<<<(768 * 256 + 255) / 256, 256, 0, stream>>>(wl_in,  wl_in_h,  768 * 256);
  k_f32_to_f16<<<(256 * 256 + 255) / 256, 256, 0, stream>>>(wl_out, wl_out_h, 256 * 256);
  k_f32_to_f16<<<(768 * 256 + 255) / 256, 256, 0, stream>>>(ws_in,  ws_in_h,  768 * 256);
  k_f32_to_f16<<<(256 * 256 + 255) / 256, 256, 0, stream>>>(ws_out, ws_out_h, 256 * 256);

  const int TC_BLOCKS = (TOK * CDIM) / 256;   // 16384
  // stage-1 inputs: B0=Xq1, B1=Xk1, B2=Xv1, B3=Pl
  k_pack_long<<<TC_BLOCKS, 256, 0, stream>>>(q, k, v, pos, B[0], B[1], B[2], B[3]);

  dim3 g256(256 / 64, TOK / 64);              // N=256 GEMMs
  // in-proj 1: Q1=B4, K1=B5, V1=B6
  k_gemm<h16><<<g256, 256, 0, stream>>>(B[0], wl_in_h,             bl_in,       B[4], TOK, 256, 256);
  k_gemm<h16><<<g256, 256, 0, stream>>>(B[1], wl_in_h + 256 * 256, bl_in + 256, B[5], TOK, 256, 256);
  k_gemm<h16><<<g256, 256, 0, stream>>>(B[2], wl_in_h + 512 * 256, bl_in + 512, B[6], TOK, 256, 256);

  // attention 1: O1=B7   (4096 problems, 2 waves/block)
  k_attn1<<<2048, 64, 0, stream>>>(B[4], B[5], B[6], B[7]);

  // out-proj 1: v1 = B8
  k_gemm<h16><<<g256, 256, 0, stream>>>(B[7], wl_out_h, bl_out, B[8], TOK, 256, 256);

  // long->short: Xq2=B0 (v1+pos), Xv2=B1
  k_long_to_short<<<TC_BLOCKS, 256, 0, stream>>>(B[8], B[3], B[0], B[1]);

  // in-proj 2: Q2=B2, K2=B3, V2=B4 (q and k share input Xq2)
  k_gemm<h16><<<g256, 256, 0, stream>>>(B[0], ws_in_h,             bs_in,       B[2], TOK, 256, 256);
  k_gemm<h16><<<g256, 256, 0, stream>>>(B[0], ws_in_h + 256 * 256, bs_in + 256, B[3], TOK, 256, 256);
  k_gemm<h16><<<g256, 256, 0, stream>>>(B[1], ws_in_h + 512 * 256, bs_in + 512, B[4], TOK, 256, 256);

  // attention 2: O2=B5   (32 b2 * 8 heads * 32 query-blocks)
  k_attn2<<<32 * 8 * 32, 128, 0, stream>>>(B[2], B[3], B[4], B[5]);

  // out-proj 2 -> f32 short-layout Y2
  k_gemm<float><<<g256, 256, 0, stream>>>(B[5], ws_out_h, bs_out, Y2, TOK, 256, 256);

  // scatter to [n,c,t,h,w]
  k_scatter<<<TC_BLOCKS, 256, 0, stream>>>(Y2, out);
}
